// DeepSetModel_11759620456594
// MI455X (gfx1250) — matrix operand order repaired and checked by reading, not compiled, not run on hardware
//
#include <hip/hip_runtime.h>
#include <hip/hip_bf16.h>
#include <math.h>

// ---------------------------------------------------------------------------
// DeepSet forward, fused for MI455X (gfx1250, wave32, WMMA 16x16x32 f16).
//
//   phi:  h = relu(relu(x@W1+b1)@W2+b2)        [2M,32] -> [2M,64]
//   pool: pooled[eid] += h   (event_ids sorted, all ids present => compacted
//                             segment index == event_id itself)
//   rho:  out = sigmoid(relu(relu(pooled@R1+c1)@R2+c2)@R3+c3)  [100K,64]->[100K,1]
//
// Strategy: bandwidth-bound fused phi+pool (single pass over x, run-length-
// compressed f32 atomics into pooled in d_ws), then a small rho kernel.
// All GEMMs via v_wmma_f32_16x16x32_f16 (f32 accumulate).
// ---------------------------------------------------------------------------

typedef __attribute__((ext_vector_type(16))) _Float16 v16h;
typedef __attribute__((ext_vector_type(8)))  float    v8f;

#define N_TRACKS 2000000
#define N_EVENTS 100000
#define N_FEAT   32
#define N_LATENT 64

__device__ __forceinline__ v8f wmma_f16(v16h a, v16h b, v8f c) {
  // (neg_a, A, neg_b, B, c_mod, C, reuse_a, reuse_b)
  return __builtin_amdgcn_wmma_f32_16x16x32_f16(false, a, false, b, (short)0, c,
                                                false, false);
}

// A fragment (16x32, f16) from row-major f32 source, rows [row0,row0+16).
// Lane layout (ISA 7.12.2, 16-bit A 16x32): m = lane&15; lanes 0-15 hold
// K in {8h..8h+7} U {16+8h..16+8h+7} with h = lane>>4.
__device__ __forceinline__ v16h load_a_f32(const float* __restrict__ X, int ld,
                                           int row0, int k0, int lane) {
  int m  = row0 + (lane & 15);
  int h8 = (lane >> 4) * 8;
  const float* r = X + (long)m * ld + k0;
  v16h a;
#pragma unroll
  for (int j = 0; j < 8; ++j) a[j]     = (_Float16)r[h8 + j];
#pragma unroll
  for (int j = 0; j < 8; ++j) a[8 + j] = (_Float16)r[16 + h8 + j];
  return a;
}

// Same, but with the row index clamped (tail tiles in rho).
__device__ __forceinline__ v16h load_a_f32_clamp(const float* __restrict__ X, int ld,
                                                 int row0, int k0, int lane, int nrow) {
  int m = row0 + (lane & 15);
  m = m < nrow ? m : nrow - 1;
  int h8 = (lane >> 4) * 8;
  const float* r = X + (long)m * ld + k0;
  v16h a;
#pragma unroll
  for (int j = 0; j < 8; ++j) a[j]     = (_Float16)r[h8 + j];
#pragma unroll
  for (int j = 0; j < 8; ++j) a[8 + j] = (_Float16)r[16 + h8 + j];
  return a;
}

// A fragment from an LDS f16 tile (row-major, leading dim ld).
__device__ __forceinline__ v16h load_a_lds(const _Float16* __restrict__ H, int ld,
                                           int k0, int lane) {
  int m  = lane & 15;
  int h8 = (lane >> 4) * 8;
  const _Float16* r = H + m * ld + k0;
  v16h a;
#pragma unroll
  for (int j = 0; j < 8; ++j) a[j]     = r[h8 + j];
#pragma unroll
  for (int j = 0; j < 8; ++j) a[8 + j] = r[16 + h8 + j];
  return a;
}

// B fragment (32x16, f16) from row-major f32 weight W[K x N].
// Lane layout: n = lane&15; K in {8h..8h+7} U {16+8h..16+8h+7}, h = lane>>4.
__device__ __forceinline__ v16h load_b_f32(const float* __restrict__ W, int N,
                                           int k0, int n0, int lane) {
  int n  = n0 + (lane & 15);
  int kh = k0 + (lane >> 4) * 8;
  v16h b;
#pragma unroll
  for (int j = 0; j < 16; ++j) b[j] = (_Float16)W[(kh + (j & 7) + (j >> 3) * 16) * N + n];
  return b;
}

// C tile (f32 16x16) -> bias + relu -> LDS f16 tile (row-major).
// C layout: lane holds column n0+(lane&15); rows v + 8*(lane>>4) in c[v].
__device__ __forceinline__ void store_c_relu_lds(v8f c, float bias,
                                                 _Float16* __restrict__ H, int ld,
                                                 int n0, int lane) {
  int col = n0 + (lane & 15);
  int r0  = (lane >> 4) * 8;
#pragma unroll
  for (int v = 0; v < 8; ++v) {
    float t = fmaxf(c[v] + bias, 0.f);
    H[(r0 + v) * ld + col] = (_Float16)t;
  }
}

// ---------------------------------------------------------------------------
// Kernel 1: zero the pooled accumulator (d_ws is poisoned, must re-init).
// ---------------------------------------------------------------------------
__global__ void zero_kernel(float* __restrict__ p, int n) {
  int stride = gridDim.x * blockDim.x;
  for (int i = blockIdx.x * blockDim.x + threadIdx.x; i < n; i += stride)
    p[i] = 0.f;
}

// ---------------------------------------------------------------------------
// Kernel 2: fused phi MLP + segment-sum pooling.
// One wave = one 16-track tile per loop iteration (grid-stride, persistent).
// ---------------------------------------------------------------------------
__global__ void __launch_bounds__(256)
phi_pool_kernel(const float* __restrict__ x, const long long* __restrict__ eids,
                const float* __restrict__ w1, const float* __restrict__ b1,
                const float* __restrict__ w2, const float* __restrict__ b2,
                float* __restrict__ pooled) {
  __shared__ _Float16 lds[8][16 * 64];
  const int lane = threadIdx.x & 31;
  const int wave = threadIdx.x >> 5;
  _Float16* H = lds[wave];

  // Weights -> f16 B fragments, held in registers for the whole wave lifetime.
  v16h B1[4];                 // W1: 32x64  -> 1 K-chunk x 4 N-tiles
#pragma unroll
  for (int n = 0; n < 4; ++n) B1[n] = load_b_f32(w1, 64, 0, n * 16, lane);
  v16h B2[2][4];              // W2: 64x64  -> 2 K-chunks x 4 N-tiles
#pragma unroll
  for (int k = 0; k < 2; ++k)
#pragma unroll
    for (int n = 0; n < 4; ++n) B2[k][n] = load_b_f32(w2, 64, k * 32, n * 16, lane);

  float b1c[4], b2c[4];
#pragma unroll
  for (int n = 0; n < 4; ++n) { b1c[n] = b1[n * 16 + (lane & 15)];
                                b2c[n] = b2[n * 16 + (lane & 15)]; }

  const int waveId = blockIdx.x * 8 + wave;
  const int nWaves = gridDim.x * 8;
  const int nTiles = N_TRACKS / 16;       // 125000, exact

  for (int tile = waveId; tile < nTiles; tile += nWaves) {
    const int t0 = tile * 16;

    // Prefetch the next tile of x this wave will touch (global_prefetch_b8).
    if (tile + nWaves < nTiles)
      __builtin_prefetch(x + (long)(t0 + nWaves * 16) * N_FEAT, 0, 0);

    // ---- layer 1: [16x32] @ [32x64] ----
    v16h a = load_a_f32(x, N_FEAT, t0, 0, lane);
#pragma unroll
    for (int n = 0; n < 4; ++n) {
      v8f c = {};
      c = wmma_f16(a, B1[n], c);
      store_c_relu_lds(c, b1c[n], H, 64, n * 16, lane);   // relu(h1) -> LDS (f16)
    }
    // LDS is in-order per wave; just drain DScnt before the transposed re-read.
    asm volatile("s_wait_dscnt 0" ::: "memory");

    v16h a2lo = load_a_lds(H, 64, 0, lane);
    v16h a2hi = load_a_lds(H, 64, 32, lane);

    // event ids for this lane's 8 consecutive rows (sorted -> runs)
    const int r0 = t0 + (lane >> 4) * 8;
    int ev[8];
#pragma unroll
    for (int v = 0; v < 8; ++v) ev[v] = (int)eids[r0 + v];

    // ---- layer 2 + run-length-compressed segment atomics ----
#pragma unroll
    for (int n = 0; n < 4; ++n) {
      v8f c = {};
      c = wmma_f16(a2lo, B2[0][n], c);
      c = wmma_f16(a2hi, B2[1][n], c);
      const int col = n * 16 + (lane & 15);
      float acc = fmaxf(c[0] + b2c[n], 0.f);
      int cur = ev[0];
#pragma unroll
      for (int v = 1; v < 8; ++v) {
        float t = fmaxf(c[v] + b2c[n], 0.f);
        if (ev[v] == cur) {
          acc += t;
        } else {
          atomicAdd(&pooled[cur * N_LATENT + col], acc);
          acc = t;
          cur = ev[v];
        }
      }
      atomicAdd(&pooled[cur * N_LATENT + col], acc);
    }
  }
}

// ---------------------------------------------------------------------------
// Kernel 3: rho MLP + sigmoid. One wave = one 16-event tile.
// ---------------------------------------------------------------------------
__global__ void __launch_bounds__(256)
rho_kernel(const float* __restrict__ pooled,
           const float* __restrict__ w1, const float* __restrict__ b1,
           const float* __restrict__ w2, const float* __restrict__ b2,
           const float* __restrict__ w3, const float* __restrict__ b3p,
           float* __restrict__ out) {
  __shared__ _Float16 lds[8][16 * 128];
  const int lane = threadIdx.x & 31;
  const int wave = threadIdx.x >> 5;
  _Float16* H = lds[wave];

  float b1c[8], b2c[4], w3c[4];
#pragma unroll
  for (int n = 0; n < 8; ++n) b1c[n] = b1[n * 16 + (lane & 15)];
#pragma unroll
  for (int n = 0; n < 4; ++n) { b2c[n] = b2[n * 16 + (lane & 15)];
                                w3c[n] = w3[n * 16 + (lane & 15)]; }
  const float b3 = b3p[0];

  const int waveId = blockIdx.x * 8 + wave;
  const int nWaves = gridDim.x * 8;
  const int nTiles = (N_EVENTS + 15) / 16;   // 6250

  for (int tile = waveId; tile < nTiles; tile += nWaves) {
    const int e0 = tile * 16;

    // ---- layer 1: [16x64] @ [64x128] ----
    v16h a0 = load_a_f32_clamp(pooled, 64, e0, 0,  lane, N_EVENTS);
    v16h a1 = load_a_f32_clamp(pooled, 64, e0, 32, lane, N_EVENTS);
#pragma unroll
    for (int n = 0; n < 8; ++n) {
      v8f c = {};
      c = wmma_f16(a0, load_b_f32(w1, 128, 0,  n * 16, lane), c);
      c = wmma_f16(a1, load_b_f32(w1, 128, 32, n * 16, lane), c);
      store_c_relu_lds(c, b1c[n], H, 128, n * 16, lane);
    }
    asm volatile("s_wait_dscnt 0" ::: "memory");

    v16h a2[4];
#pragma unroll
    for (int k = 0; k < 4; ++k) a2[k] = load_a_lds(H, 128, k * 32, lane);

    // ---- layer 2: [16x128] @ [128x64], fused with the 64->1 dot (w3) ----
    float partial[8];
#pragma unroll
    for (int v = 0; v < 8; ++v) partial[v] = 0.f;
#pragma unroll
    for (int n = 0; n < 4; ++n) {
      v8f c = {};
#pragma unroll
      for (int k = 0; k < 4; ++k)
        c = wmma_f16(a2[k], load_b_f32(w2, 64, k * 32, n * 16, lane), c);
#pragma unroll
      for (int v = 0; v < 8; ++v) {
        float t = fmaxf(c[v] + b2c[n], 0.f);        // relu(r2)
        partial[v] = fmaf(t, w3c[n], partial[v]);   // * w3[col], accumulate
      }
    }
    // Reduce across the 16 columns held by each half-wave (lanes 0-15: rows
    // 0-7; lanes 16-31: rows 8-15). xor masks 1,2,4,8 stay inside the group.
#pragma unroll
    for (int off = 1; off < 16; off <<= 1)
#pragma unroll
      for (int v = 0; v < 8; ++v)
        partial[v] += __shfl_xor(partial[v], off, 32);

    if ((lane & 15) == 0) {
      const int rbase = (lane >> 4) * 8;
#pragma unroll
      for (int v = 0; v < 8; ++v) {
        int e = e0 + rbase + v;
        if (e < N_EVENTS)
          out[e] = 1.f / (1.f + __expf(-(partial[v] + b3)));
      }
    }
  }
}

// ---------------------------------------------------------------------------
extern "C" void kernel_launch(void* const* d_in, const int* in_sizes, int n_in,
                              void* d_out, int out_size, void* d_ws, size_t ws_size,
                              hipStream_t stream) {
  const float*     x      = (const float*)d_in[0];
  const long long* eids   = (const long long*)d_in[1];   // int64 event ids
  const float*     phi_w1 = (const float*)d_in[2];
  const float*     phi_b1 = (const float*)d_in[3];
  const float*     phi_w2 = (const float*)d_in[4];
  const float*     phi_b2 = (const float*)d_in[5];
  const float*     rho_w1 = (const float*)d_in[6];
  const float*     rho_b1 = (const float*)d_in[7];
  const float*     rho_w2 = (const float*)d_in[8];
  const float*     rho_b2 = (const float*)d_in[9];
  const float*     rho_w3 = (const float*)d_in[10];
  const float*     rho_b3 = (const float*)d_in[11];
  float*           out    = (float*)d_out;

  float* pooled = (float*)d_ws;            // [N_EVENTS, N_LATENT] f32 = 25.6 MB
  const int pooledN = N_EVENTS * N_LATENT;

  zero_kernel<<<2048, 256, 0, stream>>>(pooled, pooledN);
  phi_pool_kernel<<<1024, 256, 0, stream>>>(x, eids, phi_w1, phi_b1,
                                            phi_w2, phi_b2, pooled);
  rho_kernel<<<256, 256, 0, stream>>>(pooled, rho_w1, rho_b1, rho_w2, rho_b2,
                                      rho_w3, rho_b3, out);
}
